// Net_49950469652573
// MI455X (gfx1250) — matrix operand order repaired and checked by reading, not compiled, not run on hardware
//
#include <hip/hip_runtime.h>
#include <math.h>

typedef __attribute__((ext_vector_type(16))) __bf16 v16bf;
typedef __attribute__((ext_vector_type(8)))  __bf16 v8bf;
typedef __attribute__((ext_vector_type(8)))  float  v8f;

// K reordering: k' = s*160 + cr, cr = c*9 + r in [0,153) padded to 160 (zeros).
// 45 K-steps of 32; each step lies within a single s (160 = 5*32).
#define KS2 45
#define HSTRIDE 168   // padded cr-stride of transposed halo: 84 dwords -> conflict-free b128

__device__ __forceinline__ float sigm(float x) { return 1.0f / (1.0f + expf(-x)); }

// ---------------------------------------------------------------- zero init
__global__ void k_zero(float* __restrict__ p, int n) {
    int i = blockIdx.x * 256 + threadIdx.x;
    if (i < n) p[i] = 0.0f;
}

// ------------------------------------------- pack conv weights into WMMA A-fragment layout
// A-matrix 16x32 bf16 lane layout used here (contiguous k, B uses the same): lane L row M=L&15,
// element e -> K_local = ((L>>4)<<4) + e.  K order is the s-major k'.
__global__ void k_pack_w(const float* __restrict__ cw, __bf16* __restrict__ wp) {
    int mt = blockIdx.x / KS2, ks = blockIdx.x % KS2;
    int s = ks / 5, cr0 = (ks % 5) * 32;
    int lane = threadIdx.x;
    int m = mt * 16 + (lane & 15);
    __bf16* dst = wp + ((mt * KS2 + ks) * 32 + lane) * 16;
#pragma unroll
    for (int e = 0; e < 16; ++e) {
        int kl = ((lane >> 4) << 4) + e;
        int cr = cr0 + kl;
        int c = cr / 9, r = cr - c * 9;
        float v = (cr < 153) ? cw[m * 1377 + c * 81 + r * 9 + s] : 0.0f;
        dst[e] = (__bf16)v;
    }
}

// ------------------------------------------- SConv2dLSTM step: implicit-im2col bf16 WMMA GEMM
// grid 512 (one WG per (b,h) image row = 64 pixels), block 256 = 8 waves.
// Wave layout: mt = wid>>1 (16 gate channels), nt2 = wid&1 (pair of 16-col N-subtiles).
// Transposed halo haloT[x][cr]: a lane's 16 fragment elements (consecutive cr) are
// 32 contiguous bytes -> 2x ds_load_b128 per fragment, no packing moves, no k-loop barriers.
__global__ void __launch_bounds__(256) k_conv_lstm(
    const float* __restrict__ xt,   const float* __restrict__ convb,
    const __bf16* __restrict__ wp,
    const float* __restrict__ m1in, float* __restrict__ m1out,
    float* __restrict__ syn1)
{
    // haloT [72 x][168 crpad] bf16 = 24192 B; gates [64][64] f32 (16KB) aliases it after k-loop
    __shared__ __align__(128) char smem[72 * HSTRIDE * 2];
    __bf16* haloT = (__bf16*)smem;
    float*  gates = (float*)smem;

    const int tid  = threadIdx.x;
    const int lane = tid & 31, wid = tid >> 5;
    const int mt = wid >> 1, nt2 = wid & 1;

    const int b = blockIdx.x >> 6;
    const int h = blockIdx.x & 63;
    const float* xb = xt + b * 4096;
    const float* mb = m1in + b * 16 * 4096;

    // ---- one-shot receptive-field halo load, transposed (branch-free: clamp + select) ----
    for (int idx = tid; idx < 72 * 160; idx += 256) {
        int x = idx / 160, cr = idx - x * 160;
        int c = cr / 9, r = cr - c * 9;
        int hh = h + r - 4, ww = x - 4;
        bool ok = (cr < 153) && (hh >= 0) && (hh < 64) && (ww >= 0) && (ww < 64);
        int hc = min(max(hh, 0), 63), wc = min(max(ww, 0), 63);
        int cc = min(c, 16);
        const float* src = (cc == 0) ? (xb + hc * 64 + wc)
                                     : (mb + (cc - 1) * 4096 + hc * 64 + wc);
        float v = *src;
        haloT[x * HSTRIDE + cr] = (__bf16)(ok ? v : 0.0f);
    }

    v8f acc0 = {}, acc1 = {};
    __syncthreads();                     // halo ready

    const int bcol = nt2 * 32 + (lane & 15);
    // element value for fragment slot = haloT[(n+s)*HSTRIDE + cr]; linear bookkeeping only
    int hbase = bcol * HSTRIDE + (lane >> 4) * 16;   // + s*HSTRIDE + cr0 accumulated in loop
    const __bf16* wptr = wp + (mt * KS2) * 512 + lane * 16;

    for (int s9 = 0; s9 < 9; ++s9) {
        for (int q = 0; q < 5; ++q) {
            v16bf a = *(const v16bf*)wptr;
            const __bf16* hb0 = haloT + hbase;
            const __bf16* hb1 = hb0 + 16 * HSTRIDE;          // column bcol+16
            v8bf b0l = *(const v8bf*)hb0, b0h = *(const v8bf*)(hb0 + 8);
            v8bf b1l = *(const v8bf*)hb1, b1h = *(const v8bf*)(hb1 + 8);
            v16bf bf0 = __builtin_shufflevector(b0l, b0h, 0,1,2,3,4,5,6,7,8,9,10,11,12,13,14,15);
            v16bf bf1 = __builtin_shufflevector(b1l, b1h, 0,1,2,3,4,5,6,7,8,9,10,11,12,13,14,15);
            __builtin_prefetch(wptr + 512, 0, 1);
            acc0 = __builtin_amdgcn_wmma_f32_16x16x32_bf16(false, a, false, bf0,
                                                           (short)0, acc0, false, false);
            acc1 = __builtin_amdgcn_wmma_f32_16x16x32_bf16(false, a, false, bf1,
                                                           (short)0, acc1, false, false);
            wptr  += 512;                 // next A fragment block
            hbase += 32;                  // cr0 += 32
        }
        hbase += HSTRIDE - 160;           // s += 1 (x shifts by one column), cr0 -> 0
    }
    __syncthreads();                     // all halo reads done; region becomes gates

    // ---- spill gates (C/D layout: VGPR r -> M = r + (lane>=16?8:0), N = lane&15) ----
    const int mbase = mt * 16 + ((lane >> 4) << 3);
    const int nc0   = nt2 * 32 + (lane & 15);
#pragma unroll
    for (int rr = 0; rr < 8; ++rr) {
        float bias = convb[mbase + rr];
        gates[(mbase + rr) * 64 + nc0]      = acc0[rr] + bias;
        gates[(mbase + rr) * 64 + nc0 + 16] = acc1[rr] + bias;
    }
    __syncthreads();

    // ---- fused LSTM pointwise update (gate order: gi, gf, go, gg) ----
#pragma unroll
    for (int j = 0; j < 4; ++j) {
        int o = tid + j * 256;           // 0..1023 = 16 ch x 64 pixels
        int ch = o >> 6, n = o & 63;
        float gi = gates[ch * 64 + n];
        float gf = gates[(ch + 16) * 64 + n];
        float go = gates[(ch + 32) * 64 + n];
        float gg = gates[(ch + 48) * 64 + n];
        int idx = (b * 16 + ch) * 4096 + h * 64 + n;
        float sy = syn1[idx];
        sy = sigm(gf) * sy + sigm(gi) * tanhf(gg);
        syn1[idx]  = sy;
        m1out[idx] = sigm(go) * tanhf(sy);
    }
}

// ------------------------------------------- 2x2 maxpool + STE spike (131072 elems)
__global__ void k_pool_spike(const float* __restrict__ m1, float* __restrict__ spk) {
    int g = blockIdx.x * 256 + threadIdx.x;
    int b = g >> 14, rem = g & 16383;
    int ch = rem >> 10, pp = rem & 1023;
    int hp = pp >> 5, wq = pp & 31;
    const float* p = m1 + (b * 16 + ch) * 4096 + hp * 128 + wq * 2;
    float m = fmaxf(fmaxf(p[0], p[1]), fmaxf(p[64], p[65]));
    spk[g] = ((m - 1.0f) > 0.0f) ? 1.0f : 0.0f;
}

// ------------------------------------------- fc1 + Leaky (mem2) : one WG per (b, sensor)
__global__ void __launch_bounds__(256) k_fc_leaky(
    const float* __restrict__ spk, const float* __restrict__ w,
    const float* __restrict__ bias, float* __restrict__ mem2, float* __restrict__ spk2)
{
    __shared__ float red[256];
    int bi = blockIdx.x;              // 0..599 = b*75+i
    int b = bi / 75, i = bi - b * 75;
    const float* sb = spk + b * 16384;
    const float* wr = w + i * 16384;
    float p = 0.0f;
    for (int j = threadIdx.x; j < 16384; j += 256) p += sb[j] * wr[j];
    red[threadIdx.x] = p;
    __syncthreads();
    for (int s = 128; s > 0; s >>= 1) {
        if (threadIdx.x < s) red[threadIdx.x] += red[threadIdx.x + s];
        __syncthreads();
    }
    if (threadIdx.x == 0) {
        float cur = red[0] + bias[i];
        float mo  = mem2[bi];
        float mn  = 0.9f * mo + cur - ((mo > 1.0f) ? 1.0f : 0.0f);
        mem2[bi] = mn;
        spk2[bi] = ((mn - 1.0f) > 0.0f) ? 1.0f : 0.0f;
    }
}

// ------------------------------------------- tiny CfC chain, one wave, scan over batch
__device__ __forceinline__ float cfc_row(const float* W, const float* Bb, const int* M,
                                         const float* x, int len, int hh, int r)
{
    float d0 = 0, d1 = 0, d2 = 0, d3 = 0;
    const float* w0 = W + r * len;
    const float* w1 = W + (hh + r) * len;
    const float* w2 = W + (2 * hh + r) * len;
    const float* w3 = W + (3 * hh + r) * len;
    const int* mr = M + r * len;
    for (int c = 0; c < len; ++c) {
        float xv = x[c], mf = (float)mr[c];
        d0 += w0[c] * mf * xv;
        d1 += w1[c] * mf * xv;
        d2 += w2[c] * xv;
        d3 += w3[c] * xv;
    }
    float ff1 = tanhf(d0 + Bb[r]);
    float ff2 = tanhf(d1 + Bb[hh + r]);
    float ti  = sigm((d2 + Bb[2 * hh + r]) + (d3 + Bb[3 * hh + r]));
    return ff1 * (1.0f - ti) + ti * ff2;
}

__global__ void k_cfc(const float* __restrict__ spk2,
                      const float* w0, const float* b0, const float* w1, const float* b1,
                      const float* w2, const float* b2,
                      const int* m0, const int* m1, const int* m2,
                      float* __restrict__ mem3, float* __restrict__ out, int t)
{
    __shared__ float x0[84], x1[14], x2[11];
    __shared__ float h0[9],  h1[5],  h2[6];
    __shared__ float t0[9],  t1[5],  t2[6];
    int tid = threadIdx.x;
    if (tid < 9) h0[tid] = 0.0f;
    if (tid < 5) h1[tid] = 0.0f;
    if (tid < 6) h2[tid] = 0.0f;
    __syncthreads();
    for (int b = 0; b < 8; ++b) {
        for (int c = tid; c < 84; c += 32) x0[c] = (c < 75) ? spk2[b * 75 + c] : h0[c - 75];
        __syncthreads();
        if (tid < 9) t0[tid] = cfc_row(w0, b0, m0, x0, 84, 9, tid);
        __syncthreads();
        if (tid < 9)  h0[tid] = t0[tid];
        if (tid < 14) x1[tid] = (tid < 9) ? t0[tid] : h1[tid - 9];
        __syncthreads();
        if (tid < 5) t1[tid] = cfc_row(w1, b1, m1, x1, 14, 5, tid);
        __syncthreads();
        if (tid < 5)  h1[tid] = t1[tid];
        if (tid < 11) x2[tid] = (tid < 5) ? t1[tid] : h2[tid - 5];
        __syncthreads();
        if (tid < 6) t2[tid] = cfc_row(w2, b2, m2, x2, 11, 6, tid);
        __syncthreads();
        if (tid < 6) {
            h2[tid] = t2[tid];
            float mo = mem3[b * 6 + tid];
            float mn = 0.9f * mo + t2[tid] - ((mo > 1.0f) ? 1.0f : 0.0f);
            mem3[b * 6 + tid] = mn;
            out[t * 48 + b * 6 + tid]       = ((mn - 1.0f) > 0.0f) ? 1.0f : 0.0f;
            out[768 + t * 48 + b * 6 + tid] = mn;
        }
        __syncthreads();
    }
}

// ---------------------------------------------------------------- launcher
extern "C" void kernel_launch(void* const* d_in, const int* in_sizes, int n_in,
                              void* d_out, int out_size, void* d_ws, size_t ws_size,
                              hipStream_t stream)
{
    const float* x      = (const float*)d_in[0];
    const float* conv_w = (const float*)d_in[1];
    const float* conv_b = (const float*)d_in[2];
    const float* fc1_w  = (const float*)d_in[3];
    const float* fc1_b  = (const float*)d_in[4];
    const float* cfc_w0 = (const float*)d_in[5];
    const float* cfc_b0 = (const float*)d_in[6];
    const float* cfc_w1 = (const float*)d_in[7];
    const float* cfc_b1 = (const float*)d_in[8];
    const float* cfc_w2 = (const float*)d_in[9];
    const float* cfc_b2 = (const float*)d_in[10];
    const int*   mask0  = (const int*)d_in[11];
    const int*   mask1  = (const int*)d_in[12];
    const int*   mask2  = (const int*)d_in[13];
    float* out = (float*)d_out;

    // workspace layout (floats); state region [0, 1049280) is zeroed every call
    float* f     = (float*)d_ws;
    float* syn1  = f;                    // 524288
    float* m1a   = f + 524288;           // 524288 (mem1 ping)
    float* mem2  = f + 1048576;          // 600 (+pad)
    float* mem3  = f + 1049216;          // 48 (+pad)
    float* m1b   = f + 1049280;          // 524288 (mem1 pong, fully written at t=0)
    float* spk1  = f + 1573568;          // 131072
    float* spk2  = f + 1704640;          // 600 (+pad)
    __bf16* wpack = (__bf16*)(f + 1705280); // 4*45*32*16 = 92160 bf16

    k_zero<<<4099, 256, 0, stream>>>(f, 1049280);
    k_pack_w<<<4 * KS2, 32, 0, stream>>>(conv_w, wpack);

    for (int t = 0; t < 16; ++t) {
        const float* xt    = x + (size_t)t * 32768;
        const float* m1in  = (t & 1) ? m1b : m1a;
        float*       m1out = (t & 1) ? m1a : m1b;
        k_conv_lstm<<<512, 256, 0, stream>>>(xt, conv_b, wpack, m1in, m1out, syn1);
        k_pool_spike<<<512, 256, 0, stream>>>(m1out, spk1);
        k_fc_leaky<<<600, 256, 0, stream>>>(spk1, fc1_w, fc1_b, mem2, spk2);
        k_cfc<<<1, 32, 0, stream>>>(spk2, cfc_w0, cfc_b0, cfc_w1, cfc_b1, cfc_w2, cfc_b2,
                                    mask0, mask1, mask2, mem3, out, t);
    }
    (void)in_sizes; (void)n_in; (void)out_size; (void)ws_size;
}
